// AttentionBlock_57191784514063
// MI455X (gfx1250) — compile-verified, matrix operand order checked by reading, not run on hardware
//
#include <hip/hip_runtime.h>
#include <hip/hip_bf16.h>

// ---------------------------------------------------------------------------
// Types / helpers
// ---------------------------------------------------------------------------
typedef __attribute__((ext_vector_type(16))) __bf16 v16bf;
typedef __attribute__((ext_vector_type(8)))  float  v8f;
typedef __attribute__((ext_vector_type(4)))  int    v4i;

#define DEV_INLINE __device__ __forceinline__

// CDNA5 async global->LDS copy (ASYNCcnt-tracked). The builtin's prototype is
// (int4 addrspace(1)*, int4 addrspace(3)*, imm offset, imm cpol) per the
// frontend diagnostic. Falls back to load + ds_store staging if unavailable.
#if defined(__HIP_DEVICE_COMPILE__) && defined(__gfx1250__)
#if __has_builtin(__builtin_amdgcn_global_load_async_to_lds_b128)
#define ASYNC_LDS 1
#endif
#endif

#if defined(ASYNC_LDS)
typedef __attribute__((address_space(1))) v4i g1_v4i;
typedef __attribute__((address_space(3))) v4i l3_v4i;
DEV_INLINE void async_copy16(const void* g, void* l) {
    // LDS[l .. l+15] = MEM[g .. g+15]
    __builtin_amdgcn_global_load_async_to_lds_b128((g1_v4i*)g, (l3_v4i*)l, 0, 0);
}
DEV_INLINE void async_wait0() {
#if __has_builtin(__builtin_amdgcn_s_wait_asynccnt)
    __builtin_amdgcn_s_wait_asynccnt(0);
#else
    asm volatile("s_wait_asynccnt 0x0" ::: "memory");
#endif
}
#endif

DEV_INLINE __bf16 f2bf(float f) {
    union { float f; unsigned u; } a; a.f = f;
    unsigned r = a.u + 0x7FFFu + ((a.u >> 16) & 1u);   // round-to-nearest-even
    union { unsigned short s; __bf16 b; } o;
    o.s = (unsigned short)(r >> 16);
    return o.b;
}

DEV_INLINE float wave_sum(float v) {
    #pragma unroll
    for (int o = 16; o > 0; o >>= 1) v += __shfl_xor(v, o, 32);
    return v;
}
DEV_INLINE float wave_max(float v) {
    #pragma unroll
    for (int o = 16; o > 0; o >>= 1) v = fmaxf(v, __shfl_xor(v, o, 32));
    return v;
}

// ---------------------------------------------------------------------------
// fp32 -> bf16 convert (vectorized)
// ---------------------------------------------------------------------------
__global__ __launch_bounds__(256) void cvt_f32_bf16(
    const float* __restrict__ in, __bf16* __restrict__ out, size_t n4) {
    size_t i = (size_t)blockIdx.x * 256 + threadIdx.x;
    if (i >= n4) return;
    float4 f = ((const float4*)in)[i];
    union { unsigned long long u; __bf16 b[4]; } o;
    o.b[0] = f2bf(f.x); o.b[1] = f2bf(f.y); o.b[2] = f2bf(f.z); o.b[3] = f2bf(f.w);
    ((unsigned long long*)out)[i] = o.u;
}

// ---------------------------------------------------------------------------
// fp32 [K,N] -> bf16 [N,K] transpose-convert (weights go to N-major, K-contig)
// ---------------------------------------------------------------------------
__global__ __launch_bounds__(256) void transpose_cvt(
    const float* __restrict__ in, __bf16* __restrict__ out, int K, int N) {
    __shared__ float tile[32][33];
    int n0 = blockIdx.x * 32, k0 = blockIdx.y * 32;
    int tx = threadIdx.x, ty = threadIdx.y;   // block (32, 8)
    #pragma unroll
    for (int i = ty; i < 32; i += 8)
        tile[i][tx] = in[(size_t)(k0 + i) * N + (n0 + tx)];
    __syncthreads();
    #pragma unroll
    for (int i = ty; i < 32; i += 8)
        out[(size_t)(n0 + i) * K + (k0 + tx)] = f2bf(tile[tx][i]);
}

// ---------------------------------------------------------------------------
// Generic batched WMMA GEMM:  C[m,n] = act( scale * sum_k A[m,k]*B'[n,k] + bias + res )
//   Block tile 256x128x32, 256 threads = 8 wave32 (4x2), wave tile 64x64:
//   16x v_wmma_f32_16x16x32_bf16 per K-step, fp32 accumulation.
//   A: bf16 [M,K] row-major (or fp32 if A_F32; converted while staging to LDS)
//   B: bf16 "N x K" (pre-transposed weights) or, if B_TRANS, raw [K,N] layout
//      (transposed into LDS on the fly, e.g. the V matrix).
// ---------------------------------------------------------------------------
template<bool A_F32, bool B_TRANS, bool RELU, bool HAS_BIAS, bool HAS_RES,
         bool OUT_F32, bool OUT_BF16>
__global__ __launch_bounds__(256) void gemm_wmma(
    const void* __restrict__ Aptr, int ldA, long long sAb, long long sAh,
    const __bf16* __restrict__ Bptr, int ldB, long long sBb, long long sBh,
    float* __restrict__ Cf, __bf16* __restrict__ Cb, int ldC, long long sCb, long long sCh,
    const float* __restrict__ bias,
    const float* __restrict__ Rf, int ldR, long long sRb, long long sRh,
    int M, int N, int K, int nh, float scale)
{
    (void)M; (void)N;
    __shared__ alignas(16) __bf16 As[256][40];   // +8 pad: 80B stride, 16B aligned
    __shared__ alignas(16) __bf16 Bs[128][40];

    const int tid  = threadIdx.x;
    const int lane = tid & 31;
    const int wave = tid >> 5;
    const int wm   = wave >> 1;      // 0..3  (64 rows each)
    const int wn   = wave & 1;       // 0..1  (64 cols each)
    const int hlf  = lane >> 4;      // lane half (K-half selector)
    const int lr   = lane & 15;

    const int n0 = blockIdx.x * 128;
    const int m0 = blockIdx.y * 256;
    const int zb = blockIdx.z / nh;
    const int zh = blockIdx.z % nh;

    const size_t offA = (size_t)zb * sAb + (size_t)zh * sAh;
    const size_t offB = (size_t)zb * sBb + (size_t)zh * sBh;
    const size_t offC = (size_t)zb * sCb + (size_t)zh * sCh;
    const size_t offR = (size_t)zb * sRb + (size_t)zh * sRh;

    const __bf16* Ab = (const __bf16*)Aptr + offA;
    const float*  Af = (const float*)Aptr + offA;
    const __bf16* Bb = Bptr + offB;

    v8f acc[4][4] = {};

    for (int k0 = 0; k0 < K; k0 += 32) {
        // ---- stage A tile [256 x 32] into LDS (1024 chunks of 8 bf16) ----
        #pragma unroll
        for (int it = 0; it < 4; ++it) {
            int c  = tid + it * 256;
            int r  = c >> 2;
            int cc = (c & 3) << 3;
            if constexpr (A_F32) {
                const float* src = Af + (size_t)(m0 + r) * ldA + (k0 + cc);
                float4 f0 = *(const float4*)src;
                float4 f1 = *(const float4*)(src + 4);
                union { uint4 q; __bf16 b[8]; } u;
                u.b[0] = f2bf(f0.x); u.b[1] = f2bf(f0.y);
                u.b[2] = f2bf(f0.z); u.b[3] = f2bf(f0.w);
                u.b[4] = f2bf(f1.x); u.b[5] = f2bf(f1.y);
                u.b[6] = f2bf(f1.z); u.b[7] = f2bf(f1.w);
                *(uint4*)&As[r][cc] = u.q;
            } else {
                const __bf16* src = Ab + (size_t)(m0 + r) * ldA + (k0 + cc);
#if defined(ASYNC_LDS)
                async_copy16(src, &As[r][cc]);
#else
                *(uint4*)&As[r][cc] = *(const uint4*)src;
#endif
            }
        }
        // ---- stage B tile into LDS as [N=128][K=32] ----
        if constexpr (B_TRANS) {
            #pragma unroll
            for (int it = 0; it < 2; ++it) {
                int c  = tid + it * 256;       // 32 K-rows x 16 chunks
                int r  = c >> 4;               // k row
                int cc = (c & 15) << 3;        // n offset
                union { uint4 q; __bf16 b[8]; } u;
                u.q = *(const uint4*)(Bb + (size_t)(k0 + r) * ldB + (n0 + cc));
                #pragma unroll
                for (int q = 0; q < 8; ++q) Bs[cc + q][r] = u.b[q];
            }
        } else {
            #pragma unroll
            for (int it = 0; it < 2; ++it) {
                int c  = tid + it * 256;
                int r  = c >> 2;
                int cc = (c & 3) << 3;
                const __bf16* src = Bb + (size_t)(n0 + r) * ldB + (k0 + cc);
#if defined(ASYNC_LDS)
                async_copy16(src, &Bs[r][cc]);
#else
                *(uint4*)&Bs[r][cc] = *(const uint4*)src;
#endif
            }
        }
#if defined(ASYNC_LDS)
        if constexpr (!A_F32 || !B_TRANS) async_wait0();
#endif
        __syncthreads();

        // ---- fragments (ISA 16-bit layout: e0..7 -> K=half*8.., e8..15 -> K=16+half*8..) ----
        union Frag { v16bf v; uint4 q[2]; };
        Frag fa[4], fb[4];
        #pragma unroll
        for (int i = 0; i < 4; ++i) {
            int row = wm * 64 + i * 16 + lr;
            fa[i].q[0] = *(const uint4*)&As[row][hlf * 8];
            fa[i].q[1] = *(const uint4*)&As[row][16 + hlf * 8];
        }
        #pragma unroll
        for (int j = 0; j < 4; ++j) {
            int colr = wn * 64 + j * 16 + lr;
            fb[j].q[0] = *(const uint4*)&Bs[colr][hlf * 8];
            fb[j].q[1] = *(const uint4*)&Bs[colr][16 + hlf * 8];
        }
        #pragma unroll
        for (int i = 0; i < 4; ++i)
            #pragma unroll
            for (int j = 0; j < 4; ++j)
                acc[i][j] = __builtin_amdgcn_wmma_f32_16x16x32_bf16(
                    false, fa[i].v, false, fb[j].v, (short)0, acc[i][j],
                    false, false);
        __syncthreads();
    }

    // ---- epilogue ----
    float*  Cfp = Cf ? Cf + offC : nullptr;
    __bf16* Cbp = Cb ? Cb + offC : nullptr;
    const float* Rp = Rf ? Rf + offR : nullptr;

    #pragma unroll
    for (int i = 0; i < 4; ++i) {
        #pragma unroll
        for (int j = 0; j < 4; ++j) {
            int n = n0 + wn * 64 + j * 16 + lr;
            float bv = 0.f;
            if constexpr (HAS_BIAS) bv = bias[n];
            #pragma unroll
            for (int rr = 0; rr < 8; ++rr) {
                int m = m0 + wm * 64 + i * 16 + hlf * 8 + rr;
                float val = acc[i][j][rr] * scale + bv;
                if constexpr (HAS_RES) val += Rp[(size_t)m * ldR + n];
                if constexpr (RELU)    val = fmaxf(val, 0.f);
                if constexpr (OUT_F32)  Cfp[(size_t)m * ldC + n] = val;
                if constexpr (OUT_BF16) Cbp[(size_t)m * ldC + n] = f2bf(val);
            }
        }
    }
}

// ---------------------------------------------------------------------------
// Row softmax over S=1024 (in place), one 256-thread block per row
// ---------------------------------------------------------------------------
__global__ __launch_bounds__(256) void softmax_kernel(float* __restrict__ A, int S) {
    (void)S;
    size_t row = blockIdx.x;
    float* p = A + row * (size_t)1024;
    int tid = threadIdx.x;
    float v[4];
    float mx = -3.4e38f;
    #pragma unroll
    for (int c = 0; c < 4; ++c) { v[c] = p[c * 256 + tid]; mx = fmaxf(mx, v[c]); }
    __shared__ float smx[8];
    __shared__ float ssm[8];
    mx = wave_max(mx);
    if ((tid & 31) == 0) smx[tid >> 5] = mx;
    __syncthreads();
    float m = smx[0];
    #pragma unroll
    for (int w = 1; w < 8; ++w) m = fmaxf(m, smx[w]);
    float s = 0.f;
    #pragma unroll
    for (int c = 0; c < 4; ++c) { v[c] = __expf(v[c] - m); s += v[c]; }
    s = wave_sum(s);
    if ((tid & 31) == 0) ssm[tid >> 5] = s;
    __syncthreads();
    float t = 0.f;
    #pragma unroll
    for (int w = 0; w < 8; ++w) t += ssm[w];
    float inv = 1.f / t;
    #pragma unroll
    for (int c = 0; c < 4; ++c) p[c * 256 + tid] = v[c] * inv;
}

// ---------------------------------------------------------------------------
// LayerNorm over D=1024 of (in + optional res); fp32 and/or bf16 outputs
// ---------------------------------------------------------------------------
__global__ __launch_bounds__(256) void ln_kernel(
    const float* __restrict__ in, const float* __restrict__ res,
    const float* __restrict__ g, const float* __restrict__ be,
    float* __restrict__ outf, __bf16* __restrict__ outb, int D) {
    size_t row = blockIdx.x;
    const float* a = in + row * (size_t)D;
    const float* r = res ? res + row * (size_t)D : nullptr;
    int tid = threadIdx.x;
    float v[4];
    float s = 0.f, s2 = 0.f;
    #pragma unroll
    for (int c = 0; c < 4; ++c) {
        int cc = c * 256 + tid;
        float t = a[cc] + (r ? r[cc] : 0.f);
        v[c] = t; s += t; s2 += t * t;
    }
    __shared__ float sm[8];
    __shared__ float sm2[8];
    s = wave_sum(s); s2 = wave_sum(s2);
    if ((tid & 31) == 0) { sm[tid >> 5] = s; sm2[tid >> 5] = s2; }
    __syncthreads();
    float ts = 0.f, ts2 = 0.f;
    #pragma unroll
    for (int w = 0; w < 8; ++w) { ts += sm[w]; ts2 += sm2[w]; }
    float mean = ts / (float)D;
    float var  = ts2 / (float)D - mean * mean;
    float rstd = rsqrtf(var + 1e-5f);
    #pragma unroll
    for (int c = 0; c < 4; ++c) {
        int cc = c * 256 + tid;
        float o = (v[c] - mean) * rstd * g[cc] + be[cc];
        if (outf) outf[row * (size_t)D + cc] = o;
        if (outb) outb[row * (size_t)D + cc] = f2bf(o);
    }
}

// ---------------------------------------------------------------------------
// Launcher
// ---------------------------------------------------------------------------
extern "C" void kernel_launch(void* const* d_in, const int* in_sizes, int n_in,
                              void* d_out, int out_size, void* d_ws, size_t ws_size,
                              hipStream_t stream) {
    (void)in_sizes; (void)n_in; (void)out_size; (void)ws_size;

    constexpr int Bn = 8, L = 1024, S = 1024, D = 1024, H = 8, E = 128, DFF = 4096;
    constexpr int HE = H * E;          // 1024
    constexpr int BL = Bn * L;         // 8192 rows

    const float* x    = (const float*)d_in[0];
    const float* cross= (const float*)d_in[1];
    const float* Wq   = (const float*)d_in[2];
    const float* bq   = (const float*)d_in[3];
    const float* Wk   = (const float*)d_in[4];
    const float* bk   = (const float*)d_in[5];
    const float* Wv   = (const float*)d_in[6];
    const float* bv   = (const float*)d_in[7];
    const float* Wo   = (const float*)d_in[8];
    const float* bo   = (const float*)d_in[9];
    const float* W1   = (const float*)d_in[10];
    const float* b1   = (const float*)d_in[11];
    const float* W2   = (const float*)d_in[12];
    const float* b2   = (const float*)d_in[13];
    const float* g1   = (const float*)d_in[14];
    const float* be1  = (const float*)d_in[15];
    const float* g2   = (const float*)d_in[16];
    const float* be2  = (const float*)d_in[17];

    // workspace bump allocator (256B aligned)
    char* p = (char*)d_ws;
    auto alloc = [&](size_t bytes) -> void* {
        void* r = (void*)p;
        p += (bytes + 255) & ~(size_t)255;
        return r;
    };

    __bf16* xb   = (__bf16*)alloc((size_t)BL * D * 2);
    __bf16* crb  = (__bf16*)alloc((size_t)BL * D * 2);
    __bf16* Wqt  = (__bf16*)alloc((size_t)D * HE * 2);
    __bf16* Wkt  = (__bf16*)alloc((size_t)D * HE * 2);
    __bf16* Wvt  = (__bf16*)alloc((size_t)D * HE * 2);
    __bf16* Wot  = (__bf16*)alloc((size_t)HE * D * 2);
    __bf16* W1t  = (__bf16*)alloc((size_t)D * DFF * 2);
    __bf16* W2t  = (__bf16*)alloc((size_t)DFF * D * 2);
    __bf16* Qb   = (__bf16*)alloc((size_t)BL * HE * 2);
    __bf16* Kb   = (__bf16*)alloc((size_t)BL * HE * 2);
    __bf16* Vb   = (__bf16*)alloc((size_t)BL * HE * 2);
    __bf16* AOb  = (__bf16*)alloc((size_t)BL * HE * 2);   // attention output (bf16)
    float*  newx = (float*)alloc((size_t)BL * D * 4);
    float*  x1f  = (float*)alloc((size_t)BL * D * 4);
    __bf16* x1b  = (__bf16*)alloc((size_t)BL * D * 2);
    __bf16* hb   = (__bf16*)alloc((size_t)BL * DFF * 2);
    float*  yf   = (float*)alloc((size_t)BL * D * 4);

    float* out2 = (float*)d_out;                              // [B, L, D]
    float* Aout = (float*)d_out + (size_t)Bn * L * D;         // [B, H, L, S]

    // -- convert activations to bf16 --
    {
        size_t n4 = (size_t)BL * D / 4;
        int blocks = (int)((n4 + 255) / 256);
        cvt_f32_bf16<<<blocks, 256, 0, stream>>>(x, xb, n4);
        cvt_f32_bf16<<<blocks, 256, 0, stream>>>(cross, crb, n4);
    }
    // -- transpose-convert weights to bf16 [N,K] --
    {
        dim3 tb(32, 8);
        transpose_cvt<<<dim3(HE / 32, D / 32), tb, 0, stream>>>(Wq, Wqt, D, HE);
        transpose_cvt<<<dim3(HE / 32, D / 32), tb, 0, stream>>>(Wk, Wkt, D, HE);
        transpose_cvt<<<dim3(HE / 32, D / 32), tb, 0, stream>>>(Wv, Wvt, D, HE);
        transpose_cvt<<<dim3(D / 32, HE / 32), tb, 0, stream>>>(Wo, Wot, HE, D);
        transpose_cvt<<<dim3(DFF / 32, D / 32), tb, 0, stream>>>(W1, W1t, D, DFF);
        transpose_cvt<<<dim3(D / 32, DFF / 32), tb, 0, stream>>>(W2, W2t, DFF, D);
    }

    // -- Q/K/V projections (bf16 out, +bias) --
    gemm_wmma<false, false, false, true, false, false, true>
        <<<dim3(HE / 128, BL / 256, 1), 256, 0, stream>>>(
            xb, D, 0, 0, Wqt, D, 0, 0,
            nullptr, Qb, HE, 0, 0, bq,
            nullptr, 0, 0, 0, BL, HE, D, 1, 1.0f);
    gemm_wmma<false, false, false, true, false, false, true>
        <<<dim3(HE / 128, BL / 256, 1), 256, 0, stream>>>(
            crb, D, 0, 0, Wkt, D, 0, 0,
            nullptr, Kb, HE, 0, 0, bk,
            nullptr, 0, 0, 0, BL, HE, D, 1, 1.0f);
    gemm_wmma<false, false, false, true, false, false, true>
        <<<dim3(HE / 128, BL / 256, 1), 256, 0, stream>>>(
            crb, D, 0, 0, Wvt, D, 0, 0,
            nullptr, Vb, HE, 0, 0, bv,
            nullptr, 0, 0, 0, BL, HE, D, 1, 1.0f);

    // -- scores = scale * Q K^T  (per (b,h); K already in [s, e] = N x K form) --
    gemm_wmma<false, false, false, false, false, true, false>
        <<<dim3(S / 128, L / 256, Bn * H), 256, 0, stream>>>(
            Qb, HE, (long long)L * HE, (long long)E,
            Kb, HE, (long long)S * HE, (long long)E,
            Aout, nullptr, S, (long long)H * L * S, (long long)L * S,
            nullptr, nullptr, 0, 0, 0,
            L, S, E, H, 0.0883883476483184f /* 1/sqrt(128) */);

    // -- softmax rows in place (A is part of the output) --
    softmax_kernel<<<Bn * H * L, 256, 0, stream>>>(Aout, S);

    // -- out = A @ V  (A fp32 converted while staging; V transposed into LDS) --
    gemm_wmma<true, true, false, false, false, false, true>
        <<<dim3(E / 128, L / 256, Bn * H), 256, 0, stream>>>(
            Aout, S, (long long)H * L * S, (long long)L * S,
            Vb, HE, (long long)S * HE, (long long)E,
            nullptr, AOb, HE, (long long)L * HE, (long long)E,
            nullptr, nullptr, 0, 0, 0,
            L, E, S, H, 1.0f);

    // -- new_x = out @ Wo + bo  (fp32 out) --
    gemm_wmma<false, false, false, true, false, true, false>
        <<<dim3(D / 128, BL / 256, 1), 256, 0, stream>>>(
            AOb, HE, 0, 0, Wot, HE, 0, 0,
            newx, nullptr, D, 0, 0, bo,
            nullptr, 0, 0, 0, BL, D, HE, 1, 1.0f);

    // -- x1 = LN(new_x + x) : fp32 (for residual) + bf16 (for FFN GEMM) --
    ln_kernel<<<BL, 256, 0, stream>>>(newx, x, g1, be1, x1f, x1b, D);

    // -- h = relu(x1 @ W1 + b1) (bf16 out) --
    gemm_wmma<false, false, true, true, false, false, true>
        <<<dim3(DFF / 128, BL / 256, 1), 256, 0, stream>>>(
            x1b, D, 0, 0, W1t, D, 0, 0,
            nullptr, hb, DFF, 0, 0, b1,
            nullptr, 0, 0, 0, BL, DFF, D, 1, 1.0f);

    // -- y + x1 = h @ W2 + b2 + x1 (fp32 out) --
    gemm_wmma<false, false, false, true, true, true, false>
        <<<dim3(D / 128, BL / 256, 1), 256, 0, stream>>>(
            hb, DFF, 0, 0, W2t, DFF, 0, 0,
            yf, nullptr, D, 0, 0, b2,
            x1f, D, 0, 0, BL, D, DFF, 1, 1.0f);

    // -- out2 = LN(x1 + y) --
    ln_kernel<<<BL, 256, 0, stream>>>(yf, nullptr, g2, be2, out2, nullptr, D);
}
